// ILLVGG16RoIHead_9234179687160
// MI455X (gfx1250) — matrix operand order repaired and checked by reading, not compiled, not run on hardware
//
#include <hip/hip_runtime.h>
#include <hip/hip_bf16.h>
#include <float.h>

// ---------------------------------------------------------------------------
// Types for CDNA5 WMMA (wave32): 16x16x32 bf16 -> f32, plus TDM descriptors
// ---------------------------------------------------------------------------
typedef __attribute__((ext_vector_type(16))) __bf16 v16bf;
typedef __attribute__((ext_vector_type(8)))  float  v8f;
typedef __attribute__((ext_vector_type(4)))  unsigned int u32x4;
typedef __attribute__((ext_vector_type(8)))  int i32x8;
typedef __attribute__((ext_vector_type(4)))  int i32x4;

struct U32x8 { uint4 lo, hi; };   // 32 bytes == one v16bf operand

__device__ __forceinline__ unsigned short f2bf(float f) {
  unsigned int u = __builtin_bit_cast(unsigned int, f);
  u += 0x7FFFu + ((u >> 16) & 1u);           // round-to-nearest-even
  return (unsigned short)(u >> 16);
}

// ---------------------------------------------------------------------------
// Problem constants
// ---------------------------------------------------------------------------
#define NB    4
#define CCH   512
#define HH    38
#define WW    38
#define RR    1024
#define POOL  7
#define DIN   (CCH * POOL * POOL)   // 25088
#define DH    4096
#define NCLS4 84
#define NCLS  21
#define NPAD  128                   // heads padded to one BN tile

// ---------------------------------------------------------------------------
// Kernel 1: RoI max pool, quantized bins, empty bin -> 0, bf16 output
// ---------------------------------------------------------------------------
__global__ __launch_bounds__(256)
void roi_pool_kernel(const float* __restrict__ x,
                     const float* __restrict__ rois,
                     const int*   __restrict__ roi_idx,
                     unsigned short* __restrict__ out)
{
  const int k   = blockIdx.x * 256 + threadIdx.x;   // 0..25087
  const int r   = blockIdx.y;
  const int c   = k / 49;
  const int bin = k - c * 49;
  const int ph  = bin / POOL, pw = bin - ph * POOL;

  // reference permutes rois[:, [1,0,3,2]] -> (x1,y1,x2,y2)
  const float scale = 1.0f / 16.0f;
  const int x1 = (int)rintf(rois[r * 4 + 1] * scale);
  const int y1 = (int)rintf(rois[r * 4 + 0] * scale);
  const int x2 = (int)rintf(rois[r * 4 + 3] * scale);
  const int y2 = (int)rintf(rois[r * 4 + 2] * scale);

  const int roi_w = max(x2 - x1 + 1, 1);
  const int roi_h = max(y2 - y1 + 1, 1);

  int hs = min(max((ph * roi_h) / POOL + y1, 0), HH);
  int he = min(max(((ph + 1) * roi_h + POOL - 1) / POOL + y1, 0), HH);
  int ws = min(max((pw * roi_w) / POOL + x1, 0), WW);
  int we = min(max(((pw + 1) * roi_w + POOL - 1) / POOL + x1, 0), WW);

  const int img = roi_idx[r];
  const float* feat = x + ((size_t)img * CCH + c) * (HH * WW);

  float m = -FLT_MAX;
  for (int h = hs; h < he; ++h)
    for (int w = ws; w < we; ++w)
      m = fmaxf(m, feat[h * WW + w]);
  if (m == -FLT_MAX) m = 0.0f;

  out[(size_t)r * DIN + k] = f2bf(m);
}

// ---------------------------------------------------------------------------
// Kernel 2: fp32 -> bf16 convert with optional zero column padding
// ---------------------------------------------------------------------------
__global__ __launch_bounds__(256)
void convert_pad_kernel(const float* __restrict__ src,
                        unsigned short* __restrict__ dst,
                        int rows, int scols, int dcols)
{
  const size_t total = (size_t)rows * dcols;
  for (size_t i = (size_t)blockIdx.x * blockDim.x + threadIdx.x;
       i < total; i += (size_t)gridDim.x * blockDim.x) {
    const size_t row = i / dcols;
    const int    col = (int)(i - row * dcols);
    const float  v   = (col < scols) ? src[row * scols + col] : 0.0f;
    dst[i] = f2bf(v);
  }
}

// ---------------------------------------------------------------------------
// Kernel 3: bf16 GEMM, C = relu(A * B + bias)
//   A tile staged by the Tensor Data Mover (tensor_load_to_lds), with HW row
//   padding matching LDA; B tile staged transposed by the VALU path.
// Tiles: BM=128 BN=128 BK=32; 256 threads = 8 waves in 2(M) x 4(N) grid;
// each wave computes 64x32 = 4x2 fragments via v_wmma_f32_16x16x32_bf16.
// ---------------------------------------------------------------------------
#define BM 128
#define BN 128
#define BK 32
#define LDA (BK + 8)   // 40 ushorts = 80B rows; TDM pad: every 16 dwords add 4
#define LDB (BK + 8)

__global__ __launch_bounds__(256)
void gemm_bf16(const unsigned short* __restrict__ A,
               const unsigned short* __restrict__ B,
               const float* __restrict__ bias,
               float* __restrict__ Cf,
               unsigned short* __restrict__ Cb,
               int M, int Npad, int K, int Nout, int relu)
{
  __shared__ unsigned short As[BM][LDA];   // row-major A tile (TDM-filled)
  __shared__ unsigned short Bt[BN][LDB];   // B tile stored TRANSPOSED: [n][k]

  const int tid  = threadIdx.x;
  const int lane = tid & 31;
  const int wave = tid >> 5;
  const int wm   = wave >> 2;       // 0..1 -> 64 rows each
  const int wn   = wave & 3;        // 0..3 -> 32 cols each
  const int half = lane >> 4;       // lane group
  const int l16  = lane & 15;

  const int bm0 = blockIdx.y * BM;
  const int bn0 = blockIdx.x * BN;

  // LDS byte offset of As (flat shared pointer low 32 bits == LDS offset)
  const unsigned lds_As = (unsigned)(uintptr_t)&As[0][0];

  // TDM descriptor group1 (constant across K loop):
  //   [17:16] data_size=1 (2B)  [20] pad_enable  [24:22] pad_interval=3 (16dw)
  //   [31:25] pad_amount=3 (4dw)  dim0=K dim1=M  tile=32x128  dim0_stride=K
  i32x8 g1;
  g1[0] = (int)((1u << 16) | (1u << 20) | (3u << 22) | (3u << 25));
  g1[1] = (int)(((unsigned)K & 0xFFFFu) << 16);
  g1[2] = (int)((((unsigned)K >> 16) & 0xFFFFu) | (((unsigned)M & 0xFFFFu) << 16));
  g1[3] = (int)((((unsigned)M >> 16) & 0xFFFFu) | ((unsigned)BK << 16));
  g1[4] = BM;            // tile_dim1=128, tile_dim2=0
  g1[5] = K;             // tensor_dim0_stride low 32
  g1[6] = 0;
  g1[7] = 0;
  const i32x4 gz4 = (i32x4)0;
  const i32x8 gz8 = (i32x8)0;

  v8f acc[4][2];
#pragma unroll
  for (int mt = 0; mt < 4; ++mt)
#pragma unroll
    for (int nt = 0; nt < 2; ++nt) acc[mt][nt] = (v8f)(0.0f);

  // B staging geometry: 512 chunks of 8 bf16; chunk = tid + 256*i
  const int krow0 = tid >> 4, cc0 = tid & 15;
  const int krow1 = (tid + 256) >> 4, cc1 = tid & 15;

  for (int k0 = 0; k0 < K; k0 += BK) {
    // ---- issue TDM for the A tile (one wave; EXEC-independent DMA) ----
    if (wave == 0) {
      const unsigned long long ga =
          (unsigned long long)(uintptr_t)(A + (size_t)bm0 * K + k0);
      u32x4 g0;
      g0[0] = 1u;                                  // count=1, valid D#
      g0[1] = lds_As;                              // LDS byte address
      g0[2] = (unsigned)ga;                        // global addr [31:0]
      g0[3] = ((unsigned)(ga >> 32) & 0x1FFFFFFu)  // global addr [56:32]
              | 0x80000000u;                       // type=2 (image)
      __builtin_amdgcn_tensor_load_to_lds(g0, g1, gz4, gz4, gz8, 0);
    }

    // ---- stage B tile: global b128 loads first (pipelined), then scatter ----
    const uint4 vb0 = *reinterpret_cast<const uint4*>(
        &B[(size_t)(k0 + krow0) * Npad + bn0 + cc0 * 8]);
    const uint4 vb1 = *reinterpret_cast<const uint4*>(
        &B[(size_t)(k0 + krow1) * Npad + bn0 + cc1 * 8]);
    if (k0 + BK < K)   // prefetch next B panel into GL2 (global_prefetch_b8)
      __builtin_prefetch(&B[(size_t)(k0 + BK + krow0) * Npad + bn0], 0, 1);

    const unsigned short* p0 = reinterpret_cast<const unsigned short*>(&vb0);
    const unsigned short* p1 = reinterpret_cast<const unsigned short*>(&vb1);
#pragma unroll
    for (int e = 0; e < 8; ++e) Bt[cc0 * 8 + e][krow0] = p0[e];
#pragma unroll
    for (int e = 0; e < 8; ++e) Bt[cc1 * 8 + e][krow1] = p1[e];

    __builtin_amdgcn_s_wait_tensorcnt(0);   // A tile DMA complete (wave 0)
    __syncthreads();                        // publish As + Bt to all waves

    // ---- build fragments ----
    // A (16x32 bf16): lane m=l16; e<8 -> K=half*8+e ; e>=8 -> K=16+half*8+(e-8)
    v16bf afrag[4];
#pragma unroll
    for (int mt = 0; mt < 4; ++mt) {
      const int r = wm * 64 + mt * 16 + l16;
      U32x8 t;
      t.lo = *reinterpret_cast<const uint4*>(&As[r][half * 8]);
      t.hi = *reinterpret_cast<const uint4*>(&As[r][16 + half * 8]);
      afrag[mt] = __builtin_bit_cast(v16bf, t);
    }
    // B (32x16 bf16): lane n=l16; e<8 -> K=half*8+e ; e>=8 -> K=16+half*8+(e-8)
    v16bf bfrag[2];
#pragma unroll
    for (int nt = 0; nt < 2; ++nt) {
      const int n = wn * 32 + nt * 16 + l16;
      U32x8 t;
      t.lo = *reinterpret_cast<const uint4*>(&Bt[n][half * 8]);
      t.hi = *reinterpret_cast<const uint4*>(&Bt[n][16 + half * 8]);
      bfrag[nt] = __builtin_bit_cast(v16bf, t);
    }

    // ---- 8 WMMAs per wave per K-step ----
#pragma unroll
    for (int mt = 0; mt < 4; ++mt)
#pragma unroll
      for (int nt = 0; nt < 2; ++nt)
        acc[mt][nt] = __builtin_amdgcn_wmma_f32_16x16x32_bf16(
            false, afrag[mt], false, bfrag[nt],
            (short)0, acc[mt][nt], false, false);

    __syncthreads();
  }

  // ---- epilogue: bias + relu; C/D layout: VGPR v -> row half*8+v, col l16 ----
#pragma unroll
  for (int mt = 0; mt < 4; ++mt) {
#pragma unroll
    for (int nt = 0; nt < 2; ++nt) {
      const int col = bn0 + wn * 32 + nt * 16 + l16;
      const float bv = (col < Nout) ? bias[col] : 0.0f;
#pragma unroll
      for (int v = 0; v < 8; ++v) {
        const int row = bm0 + wm * 64 + mt * 16 + half * 8 + v;
        float val = acc[mt][nt][v] + bv;
        if (relu) val = fmaxf(val, 0.0f);
        if (Cb) {
          Cb[(size_t)row * Npad + col] = f2bf(val);
        } else if (col < Nout) {
          Cf[(size_t)row * Nout + col] = val;
        }
      }
    }
  }
}

// ---------------------------------------------------------------------------
// Host-side launch
// ---------------------------------------------------------------------------
extern "C" void kernel_launch(void* const* d_in, const int* in_sizes, int n_in,
                              void* d_out, int out_size, void* d_ws, size_t ws_size,
                              hipStream_t stream) {
  const float* x    = (const float*)d_in[0];
  const float* rois = (const float*)d_in[1];
  const int*   ridx = (const int*)  d_in[2];
  const float* W1   = (const float*)d_in[3];
  const float* b1   = (const float*)d_in[4];
  const float* W2   = (const float*)d_in[5];
  const float* b2   = (const float*)d_in[6];
  const float* Wc   = (const float*)d_in[7];
  const float* bc   = (const float*)d_in[8];
  const float* Ws   = (const float*)d_in[9];
  const float* bs   = (const float*)d_in[10];
  float* out = (float*)d_out;

  // workspace layout (bf16 buffers), ~310 MB
  char* ws = (char*)d_ws;
  size_t off = 0;
  unsigned short* pool_b = (unsigned short*)(ws + off); off += (size_t)RR * DIN * 2;
  unsigned short* W1_b   = (unsigned short*)(ws + off); off += (size_t)DIN * DH * 2;
  unsigned short* fc6_b  = (unsigned short*)(ws + off); off += (size_t)RR * DH * 2;
  unsigned short* W2_b   = (unsigned short*)(ws + off); off += (size_t)DH * DH * 2;
  unsigned short* fc7_b  = (unsigned short*)(ws + off); off += (size_t)RR * DH * 2;
  unsigned short* Wc_b   = (unsigned short*)(ws + off); off += (size_t)DH * NPAD * 2;
  unsigned short* Ws_b   = (unsigned short*)(ws + off); off += (size_t)DH * NPAD * 2;

  // 1) RoI max pool -> bf16 A matrix (1024 x 25088)
  roi_pool_kernel<<<dim3(DIN / 256, RR), 256, 0, stream>>>(x, rois, ridx, pool_b);

  // 2) weight conversion (fp32 -> bf16, heads zero-padded to N=128)
  convert_pad_kernel<<<4096, 256, 0, stream>>>(W1, W1_b, DIN, DH, DH);
  convert_pad_kernel<<<2048, 256, 0, stream>>>(W2, W2_b, DH, DH, DH);
  convert_pad_kernel<<<512,  256, 0, stream>>>(Wc, Wc_b, DH, NCLS4, NPAD);
  convert_pad_kernel<<<512,  256, 0, stream>>>(Ws, Ws_b, DH, NCLS,  NPAD);

  // 3) fc6 = relu(pool @ W1 + b1)   (1024 x 4096, bf16 out)
  gemm_bf16<<<dim3(DH / BN, RR / BM), 256, 0, stream>>>(
      pool_b, W1_b, b1, nullptr, fc6_b, RR, DH, DIN, DH, 1);

  // 4) fc7 = relu(fc6 @ W2 + b2)    (1024 x 4096, bf16 out)
  gemm_bf16<<<dim3(DH / BN, RR / BM), 256, 0, stream>>>(
      fc6_b, W2_b, b2, nullptr, fc7_b, RR, DH, DH, DH, 1);

  // 5) roi_cls_locs = fc7 @ Wc + bc  (1024 x 84, fp32 -> d_out[0:86016))
  gemm_bf16<<<dim3(1, RR / BM), 256, 0, stream>>>(
      fc7_b, Wc_b, bc, out, nullptr, RR, NPAD, DH, NCLS4, 0);

  // 6) roi_scores = fc7 @ Ws + bs    (1024 x 21, fp32 -> d_out[86016:))
  gemm_bf16<<<dim3(1, RR / BM), 256, 0, stream>>>(
      fc7_b, Ws_b, bs, out + (size_t)RR * NCLS4, nullptr, RR, NPAD, DH, NCLS, 0);
}
